// BitLinear_51969104281929
// MI455X (gfx1250) — 8-bit matrix operand order repaired and checked by reading, compiled, not run on hardware
//
#include <hip/hip_runtime.h>
#include <hip/hip_bf16.h>

#define TOK   16384
#define DIN   2048
#define DOUT  2048
#define EPSF  1e-8f
#define QBF   127.0f

typedef int  v8i  __attribute__((ext_vector_type(8)));

#define WS_WFRAG   0
#define WS_XQ      (4u << 20)
#define WS_GAMMA   (36u << 20)
#define WS_XSCALE  ((36u << 20) + (16u << 10))

__global__ __launch_bounds__(256) void wquant_kernel(
    const float* __restrict__ W, unsigned int* __restrict__ wfrag,
    float* __restrict__ gamma) {
  const int n = blockIdx.x;
  const int t = threadIdx.x;
  const float* row = W + (size_t)n * DIN;

  float s = 0.f;
  for (int k = t; k < DIN; k += 256) s += fabsf(row[k]);
  __shared__ float red[256];
  red[t] = s;
  __syncthreads();
  for (int off = 128; off > 0; off >>= 1) {
    if (t < off) red[t] += red[t + off];
    __syncthreads();
  }
  const float g = red[0] * (1.0f / (float)DIN);
  if (t == 0) gamma[n] = g;
  const float inv = 1.0f / (g + EPSF);

  const int ntile = n >> 4;
  const int nlane = n & 15;

  for (int k0 = t * 8; k0 < DIN; k0 += 256 * 8) {
    const int ktile = k0 >> 6;
    const int kk    = k0 & 63;
    const int lane  = nlane + (((kk >> 4) & 1) << 4);
    const int v     = ((kk >> 5) << 2) + ((kk & 15) >> 2);
    unsigned int d0 = 0u, d1 = 0u;
#pragma unroll
    for (int b = 0; b < 4; ++b) {
      float q0 = fmaxf(-1.f, fminf(1.f, rintf(row[k0 + b] * inv)));
      float q1 = fmaxf(-1.f, fminf(1.f, rintf(row[k0 + 4 + b] * inv)));
      d0 |= ((unsigned int)(unsigned char)(signed char)(int)q0) << (8 * b);
      d1 |= ((unsigned int)(unsigned char)(signed char)(int)q1) << (8 * b);
    }
    const size_t base = (((size_t)ntile * 32 + ktile) * 32 + lane) * 8 + v;
    wfrag[base]     = d0;
    wfrag[base + 1] = d1;
  }
}

__global__ __launch_bounds__(256) void aquant_kernel(
    const float* __restrict__ X, unsigned int* __restrict__ xq,
    float* __restrict__ xscale) {
  const int m = blockIdx.x;
  const int t = threadIdx.x;
  const float* row = X + (size_t)m * DIN;

  float vals[8];
  const int k0 = t * 8;
#pragma unroll
  for (int i = 0; i < 8; ++i) vals[i] = row[k0 + i];

  float ss = 0.f, mx = 0.f;
#pragma unroll
  for (int i = 0; i < 8; ++i) {
    ss += vals[i] * vals[i];
    mx = fmaxf(mx, fabsf(vals[i]));
  }

  __shared__ float rsum[256];
  __shared__ float rmax[256];
  rsum[t] = ss;
  rmax[t] = mx;
  __syncthreads();
  for (int off = 128; off > 0; off >>= 1) {
    if (t < off) {
      rsum[t] += rsum[t + off];
      rmax[t] = fmaxf(rmax[t], rmax[t + off]);
    }
    __syncthreads();
  }
  const float rms   = sqrtf(rsum[0] * (1.0f / (float)DIN) + EPSF);
  const float maxn  = rmax[0] / rms;
  const float scl   = maxn * (1.0f / QBF);
  if (t == 0) xscale[m] = scl;
  const float qmul = 1.0f / (rms * (scl + EPSF));

  const int mtile = m >> 4;
  const int mrow  = m & 15;
  const int ktile = k0 >> 6;
  const int kk    = k0 & 63;
  const int lane  = mrow + (((kk >> 4) & 1) << 4);
  const int v     = ((kk >> 5) << 2) + ((kk & 15) >> 2);

  unsigned int d0 = 0u, d1 = 0u;
#pragma unroll
  for (int b = 0; b < 4; ++b) {
    float q0 = fmaxf(-QBF, fminf(QBF, rintf(vals[b] * qmul)));
    float q1 = fmaxf(-QBF, fminf(QBF, rintf(vals[4 + b] * qmul)));
    d0 |= ((unsigned int)(unsigned char)(signed char)(int)q0) << (8 * b);
    d1 |= ((unsigned int)(unsigned char)(signed char)(int)q1) << (8 * b);
  }
  const size_t base = (((size_t)mtile * 32 + ktile) * 32 + lane) * 8 + v;
  xq[base]     = d0;
  xq[base + 1] = d1;
}

__device__ __forceinline__ v8i ld2(const unsigned char* ubase, unsigned int off) {
  const int4 a = *(const int4*)(ubase + off);
  const int4 b = *(const int4*)(ubase + off + 16);
  return (v8i){a.x, a.y, a.z, a.w, b.x, b.y, b.z, b.w};
}

__device__ __forceinline__ v8i wmma_ss(const v8i& a, const v8i& b, const v8i& c) {
  return __builtin_amdgcn_wmma_i32_16x16x64_iu8(true, a, true, b, c, false, false);
}

__device__ __forceinline__ void store_tile(float* __restrict__ out,
                                           const v8i& acc, int mbase, int n,
                                           float g, float b,
                                           const float* xs) {
#pragma unroll
  for (int i = 0; i < 8; ++i) {
    out[(size_t)(mbase + i) * DOUT + n] = (float)acc[i] * g * xs[i] + b;
  }
}

__global__ __launch_bounds__(256) void gemm_kernel(
    const unsigned char* __restrict__ xq, const unsigned char* __restrict__ wf,
    const float* __restrict__ gamma, const float* __restrict__ xscale,
    const float* __restrict__ bias, float* __restrict__ out) {
  const int lane = threadIdx.x & 31;
  const int wave = threadIdx.x >> 5;
  const int mt0  = blockIdx.y * 2;
  const int ntb  = blockIdx.x * 32 + wave * 4;
  const int ntb_s = __builtin_amdgcn_readfirstlane(ntb);

  const unsigned char* pa = xq + (size_t)mt0 * 32768;
  const unsigned char* pb = wf + (size_t)ntb_s * 32768;
  const unsigned int lo = (unsigned int)lane * 32u;

  v8i c00 = {}, c01 = {}, c02 = {}, c03 = {};
  v8i c10 = {}, c11 = {}, c12 = {}, c13 = {};

  v8i xA0 = ld2(pa, lo);
  v8i xA1 = ld2(pa + 32768, lo);
  v8i xB0 = ld2(pb, lo);
  v8i xB1 = ld2(pb + 32768, lo);
  v8i xB2 = ld2(pb + 65536, lo);
  v8i xB3 = ld2(pb + 98304, lo);

  v8i yA0, yA1, yB0, yB1, yB2, yB3;

  for (int it = 0; it < 16; ++it) {
    yA0 = ld2(pa + 1024, lo);
    yA1 = ld2(pa + 33792, lo);
    yB0 = ld2(pb + 1024, lo);
    yB1 = ld2(pb + 33792, lo);
    yB2 = ld2(pb + 66560, lo);
    yB3 = ld2(pb + 99328, lo);

    c00 = wmma_ss(xA0, xB0, c00);
    c10 = wmma_ss(xA1, xB0, c10);
    c01 = wmma_ss(xA0, xB1, c01);
    c11 = wmma_ss(xA1, xB1, c11);
    c02 = wmma_ss(xA0, xB2, c02);
    c12 = wmma_ss(xA1, xB2, c12);
    c03 = wmma_ss(xA0, xB3, c03);
    c13 = wmma_ss(xA1, xB3, c13);

    xA0 = ld2(pa + 2048, lo);
    xA1 = ld2(pa + 34816, lo);
    xB0 = ld2(pb + 2048, lo);
    xB1 = ld2(pb + 34816, lo);
    xB2 = ld2(pb + 67584, lo);
    xB3 = ld2(pb + 100352, lo);

    c00 = wmma_ss(yA0, yB0, c00);
    c10 = wmma_ss(yA1, yB0, c10);
    c01 = wmma_ss(yA0, yB1, c01);
    c11 = wmma_ss(yA1, yB1, c11);
    c02 = wmma_ss(yA0, yB2, c02);
    c12 = wmma_ss(yA1, yB2, c12);
    c03 = wmma_ss(yA0, yB3, c03);
    c13 = wmma_ss(yA1, yB3, c13);

    pa += 2048;
    pb += 2048;
  }

  const int halfsel = lane >> 4;
  const int nlane   = lane & 15;

  float gmm[4], bs[4];
  int   nn[4];
#pragma unroll
  for (int nt = 0; nt < 4; ++nt) {
    nn[nt]  = (ntb_s + nt) * 16 + nlane;
    gmm[nt] = gamma[nn[nt]];
    bs[nt]  = bias[nn[nt]];
  }

  {
    const int mbase = mt0 * 16 + halfsel * 8;
    float xs[8];
#pragma unroll
    for (int i = 0; i < 8; ++i) xs[i] = xscale[mbase + i];
    store_tile(out, c00, mbase, nn[0], gmm[0], bs[0], xs);
    store_tile(out, c01, mbase, nn[1], gmm[1], bs[1], xs);
    store_tile(out, c02, mbase, nn[2], gmm[2], bs[2], xs);
    store_tile(out, c03, mbase, nn[3], gmm[3], bs[3], xs);
  }
  {
    const int mbase = (mt0 + 1) * 16 + halfsel * 8;
    float xs[8];
#pragma unroll
    for (int i = 0; i < 8; ++i) xs[i] = xscale[mbase + i];
    store_tile(out, c10, mbase, nn[0], gmm[0], bs[0], xs);
    store_tile(out, c11, mbase, nn[1], gmm[1], bs[1], xs);
    store_tile(out, c12, mbase, nn[2], gmm[2], bs[2], xs);
    store_tile(out, c13, mbase, nn[3], gmm[3], bs[3], xs);
  }
}

extern "C" void kernel_launch(void* const* d_in, const int* in_sizes, int n_in,
                              void* d_out, int out_size, void* d_ws, size_t ws_size,
                              hipStream_t stream) {
  const float* x    = (const float*)d_in[0];
  const float* w    = (const float*)d_in[1];
  const float* bias = (const float*)d_in[2];
  float* out        = (float*)d_out;

  char* ws = (char*)d_ws;
  unsigned int* wfrag  = (unsigned int*)(ws + WS_WFRAG);
  unsigned int* xqfrag = (unsigned int*)(ws + WS_XQ);
  float* gamma         = (float*)(ws + WS_GAMMA);
  float* xscale        = (float*)(ws + WS_XSCALE);

  wquant_kernel<<<DOUT, 256, 0, stream>>>(w, wfrag, gamma);

  aquant_kernel<<<TOK, 256, 0, stream>>>(x, xqfrag, xscale);

  gemm_kernel<<<dim3(4, TOK / 32), 256, 0, stream>>>(
      (const unsigned char*)xqfrag, (const unsigned char*)wfrag,
      gamma, xscale, bias, out);
}
